// EGTransformer_88089779241219
// MI455X (gfx1250) — matrix operand order repaired and checked by reading, not compiled, not run on hardware
//
#include <hip/hip_runtime.h>
#include <math.h>

// ---------------------------------------------------------------------------
// Swin-style transformer block for MI455X (gfx1250, wave32, WMMA).
// All GEMMs use v_wmma_f32_16x16x32_f16 (f16 in, fp32 accumulate).
// GEMM K-loop is double-buffered; if the gfx1250 async-to-LDS builtin is
// available it streams tiles with GLOBAL_LOAD_ASYNC_TO_LDS_B128 + s_wait_asynccnt,
// otherwise it falls back to register-staged double buffering.
// ---------------------------------------------------------------------------

typedef __attribute__((ext_vector_type(16))) _Float16 v16h;
typedef __attribute__((ext_vector_type(8)))  float    v8f;
typedef int v4i __attribute__((vector_size(16)));   // matches builtin param type

#define WMMA_F16(a, b, c) \
  __builtin_amdgcn_wmma_f32_16x16x32_f16(false, (a), false, (b), (short)0, (c), false, false)

#if defined(__AMDGCN__) && __has_builtin(__builtin_amdgcn_global_load_async_to_lds_b128)
#define HAVE_ASYNC_LDS 1
#else
#define HAVE_ASYNC_LDS 0
#endif

#if HAVE_ASYNC_LDS
// 16-byte async DMA: global -> LDS, tracked by ASYNCcnt.
__device__ __forceinline__ void async_cp16(void* lds, const void* gptr) {
  __builtin_amdgcn_global_load_async_to_lds_b128(
      (__attribute__((address_space(1))) v4i*)(uintptr_t)gptr,
      (__attribute__((address_space(3))) v4i*)(unsigned)(uintptr_t)lds, 0, 0);
}
__device__ __forceinline__ void async_wait0() {
  asm volatile("s_wait_asynccnt 0x0" ::: "memory");
}
#endif

// Problem constants
constexpr int   kB = 4, kH = 64, kW = 64, kC = 512, kHEADS = 8, kHD = 64;
constexpr int   kL = kH * kW;             // 4096
constexpr int   kM = kB * kL;             // 16384 token rows
constexpr int   kWH = kB * kH * kHEADS;   // 2048 window-head pairs
constexpr float kBIAS = 0.1f;
constexpr float kEPS  = 1e-5f;
constexpr float kPI_2 = 1.57079632679489662f;

// ---------------------------------------------------------------------------
// fp32 -> f16 weight conversion
// ---------------------------------------------------------------------------
__global__ __launch_bounds__(256) void cvt_kernel(const float* __restrict__ src,
                                                  _Float16* __restrict__ dst, int n) {
  int i = blockIdx.x * 256 + threadIdx.x;
  if (i < n) dst[i] = (_Float16)src[i];
}

// ---------------------------------------------------------------------------
// LayerNorm over rows of 512, output f16
// ---------------------------------------------------------------------------
__global__ __launch_bounds__(256) void ln_kernel(const float* __restrict__ x,
                                                 const float* __restrict__ g,
                                                 const float* __restrict__ b,
                                                 _Float16* __restrict__ out) {
  __shared__ float red[256];
  __shared__ float red2[256];
  const int row = blockIdx.x;
  const int t = threadIdx.x;
  const float* xr = x + (size_t)row * 512;
  float v0 = xr[t], v1 = xr[t + 256];
  red[t]  = v0 + v1;
  red2[t] = v0 * v0 + v1 * v1;
  __syncthreads();
  for (int s = 128; s > 0; s >>= 1) {
    if (t < s) { red[t] += red[t + s]; red2[t] += red2[t + s]; }
    __syncthreads();
  }
  float mean = red[0] * (1.f / 512.f);
  float var  = red2[0] * (1.f / 512.f) - mean * mean;
  float rs   = rsqrtf(var + kEPS);
  out[(size_t)row * 512 + t]       = (_Float16)((v0 - mean) * rs * g[t] + b[t]);
  out[(size_t)row * 512 + t + 256] = (_Float16)((v1 - mean) * rs * g[t + 256] + b[t + 256]);
}

// ---------------------------------------------------------------------------
// Generic WMMA GEMM: C[m,n] = sum_k A[m,k] * Bw[n,k]  (i.e. X @ W^T)
// Block tile 128x128, K step 32, double-buffered LDS. 8 wave32 per block;
// each wave owns a 32x64 subtile (2x4 fragments).
// MODE 0: f16 plain store.  MODE 1: fp32 store = resid + C + bias.
// MODE 2: f16 store = gelu(C + bias), exact erf-based GELU.
// ---------------------------------------------------------------------------
template <int MODE>
__global__ __launch_bounds__(256) void gemm_kernel(const _Float16* __restrict__ A,
                                                   const _Float16* __restrict__ Bw,
                                                   const float* __restrict__ bias,
                                                   const float* __restrict__ resid,
                                                   _Float16* __restrict__ outH,
                                                   float* __restrict__ outF,
                                                   int M, int N, int K) {
  __shared__ _Float16 As[2][128][40];   // 32 K-cols + 8 pad (row = 80 B)
  __shared__ _Float16 Bs[2][128][40];
  const int tid = threadIdx.x;
  const int lane = tid & 31, wave = tid >> 5;
  const int bm = blockIdx.y * 128, bn = blockIdx.x * 128;
  const int wm = (wave >> 1) * 32;   // 0,32,64,96
  const int wn = (wave & 1) * 64;    // 0,64
  const int rl  = lane & 15;
  const int aK0 = (lane >> 4) * 8;   // A fragment K base (CDNA5 16-bit A layout)
  const int bK0 = (lane >> 4) * 8;   // B fragment K base (same K order as A)

  // per-thread staging coordinates: 2 chunks of 16B for A and for B
  int rowc[2], colc[2];
#pragma unroll
  for (int i = 0; i < 2; i++) {
    int chunk = tid + i * 256;           // 0..511
    rowc[i] = chunk >> 2;
    colc[i] = (chunk & 3) * 8;
  }

  v8f zero = {};
  v8f acc[2][4];
#pragma unroll
  for (int i = 0; i < 2; i++)
#pragma unroll
    for (int j = 0; j < 4; j++) acc[i][j] = zero;

  const int nk = K >> 5;

#if HAVE_ASYNC_LDS
  // prologue: async-stage tile 0 into buffer 0
#pragma unroll
  for (int i = 0; i < 2; i++) {
    async_cp16(&As[0][rowc[i]][colc[i]], &A[(size_t)(bm + rowc[i]) * K + colc[i]]);
    async_cp16(&Bs[0][rowc[i]][colc[i]], &Bw[(size_t)(bn + rowc[i]) * K + colc[i]]);
  }
  async_wait0();
  __syncthreads();
#else
  // prologue: register-stage tile 0 into buffer 0
  {
    uint4 ra[2], rb[2];
#pragma unroll
    for (int i = 0; i < 2; i++) {
      ra[i] = *(const uint4*)&A[(size_t)(bm + rowc[i]) * K + colc[i]];
      rb[i] = *(const uint4*)&Bw[(size_t)(bn + rowc[i]) * K + colc[i]];
    }
#pragma unroll
    for (int i = 0; i < 2; i++) {
      *(uint4*)&As[0][rowc[i]][colc[i]] = ra[i];
      *(uint4*)&Bs[0][rowc[i]][colc[i]] = rb[i];
    }
  }
  __syncthreads();
#endif

  for (int t = 0; t < nk; t++) {
    const int buf = t & 1;
    const int ktn = (t + 1) << 5;

#if HAVE_ASYNC_LDS
    if (t + 1 < nk) {
#pragma unroll
      for (int i = 0; i < 2; i++) {
        async_cp16(&As[buf ^ 1][rowc[i]][colc[i]],
                   &A[(size_t)(bm + rowc[i]) * K + ktn + colc[i]]);
        async_cp16(&Bs[buf ^ 1][rowc[i]][colc[i]],
                   &Bw[(size_t)(bn + rowc[i]) * K + ktn + colc[i]]);
      }
    }
#else
    uint4 ra[2], rb[2];
    if (t + 1 < nk) {
#pragma unroll
      for (int i = 0; i < 2; i++) {
        ra[i] = *(const uint4*)&A[(size_t)(bm + rowc[i]) * K + ktn + colc[i]];
        rb[i] = *(const uint4*)&Bw[(size_t)(bn + rowc[i]) * K + ktn + colc[i]];
      }
    }
#endif

    // fragments + 8 WMMAs from current buffer
    v16h af[2], bf[4];
#pragma unroll
    for (int i = 0; i < 2; i++) {
      const _Float16* p = &As[buf][wm + i * 16 + rl][0];
      uint4* u = (uint4*)&af[i];
      u[0] = *(const uint4*)(p + aK0);
      u[1] = *(const uint4*)(p + aK0 + 16);
    }
#pragma unroll
    for (int j = 0; j < 4; j++) {
      const _Float16* p = &Bs[buf][wn + j * 16 + rl][0];
      uint4* u = (uint4*)&bf[j];
      u[0] = *(const uint4*)(p + bK0);
      u[1] = *(const uint4*)(p + bK0 + 16);
    }
#pragma unroll
    for (int i = 0; i < 2; i++)
#pragma unroll
      for (int j = 0; j < 4; j++) acc[i][j] = WMMA_F16(af[i], bf[j], acc[i][j]);

#if HAVE_ASYNC_LDS
    if (t + 1 < nk) async_wait0();
#else
    if (t + 1 < nk) {
#pragma unroll
      for (int i = 0; i < 2; i++) {
        *(uint4*)&As[buf ^ 1][rowc[i]][colc[i]] = ra[i];
        *(uint4*)&Bs[buf ^ 1][rowc[i]][colc[i]] = rb[i];
      }
    }
#endif
    __syncthreads();
  }

  const int mo = (lane >> 4) * 8;  // C layout: lanes 16-31 -> M+8
#pragma unroll
  for (int i = 0; i < 2; i++) {
#pragma unroll
    for (int j = 0; j < 4; j++) {
#pragma unroll
      for (int r = 0; r < 8; r++) {
        int m = bm + wm + i * 16 + mo + r;
        int n = bn + wn + j * 16 + rl;
        size_t idx = (size_t)m * N + n;
        float v = acc[i][j][r];
        if constexpr (MODE == 0) {
          outH[idx] = (_Float16)v;
        } else if constexpr (MODE == 1) {
          outF[idx] = resid[idx] + v + bias[n];
        } else {
          float tt = v + bias[n];
          outH[idx] = (_Float16)(0.5f * tt * (1.f + erff(tt * 0.70710678118654752f)));
        }
      }
    }
  }
}

// ---------------------------------------------------------------------------
// Repack QKV GEMM output (16384 x 1536 f16) into per-(window,head) tiles:
//   q[wh][xw][d], k[wh][xw][d], vT[wh][d][xw]
// ---------------------------------------------------------------------------
__global__ __launch_bounds__(256) void repack_kernel(const _Float16* __restrict__ qkv,
                                                     _Float16* __restrict__ q,
                                                     _Float16* __restrict__ k,
                                                     _Float16* __restrict__ vT) {
  int t = blockIdx.x * 256 + threadIdx.x;  // 0 .. 16384*512-1
  int rowi = t >> 9;                       // b*4096 + l
  int c = t & 511;
  int h = c >> 6, d = c & 63;
  int l = rowi & 4095, bb = rowi >> 12;
  int y = l >> 6, xw = l & 63;
  int wh = (bb * 64 + y) * 8 + h;
  size_t src = (size_t)rowi * 1536;
  q[(size_t)wh * 4096 + xw * 64 + d]  = qkv[src + c];
  k[(size_t)wh * 4096 + xw * 64 + d]  = qkv[src + 512 + c];
  vT[(size_t)wh * 4096 + d * 64 + xw] = qkv[src + 1024 + c];
}

// ---------------------------------------------------------------------------
// Pass A: per (window,head) S = q @ k^T + 0.1*rpe; store fp32 + per-block |S| sum
// ---------------------------------------------------------------------------
__global__ __launch_bounds__(256) void attn_qk_kernel(const _Float16* __restrict__ q,
                                                      const _Float16* __restrict__ k,
                                                      const float* __restrict__ rpe,
                                                      float* __restrict__ S,
                                                      float* __restrict__ partial) {
  __shared__ _Float16 qs[64][72];
  __shared__ _Float16 ks[64][72];
  __shared__ float red[256];
  const int wh = blockIdx.x, w = wh >> 3, h = wh & 7, y = w & 63;
  const int tid = threadIdx.x, lane = tid & 31, wave = tid >> 5;
  const _Float16* qp = q + (size_t)wh * 4096;
  const _Float16* kp = k + (size_t)wh * 4096;
#pragma unroll
  for (int i = 0; i < 2; i++) {
    int chunk = tid + i * 256;
    int row = chunk >> 3, cc = (chunk & 7) * 8;
    *(uint4*)&qs[row][cc] = *(const uint4*)&qp[row * 64 + cc];
    *(uint4*)&ks[row][cc] = *(const uint4*)&kp[row * 64 + cc];
  }
  __syncthreads();
  const int mbase = (wave >> 1) * 16, nbase = (wave & 1) * 32;
  const int rl = lane & 15, aK0 = (lane >> 4) * 8, bK0 = (lane >> 4) * 8;
  v8f zero = {};
  v8f acc[2];
  acc[0] = zero; acc[1] = zero;
#pragma unroll
  for (int kt = 0; kt < 64; kt += 32) {
    v16h af; uint4* ua = (uint4*)&af;
    const _Float16* pa = &qs[mbase + rl][kt];
    ua[0] = *(const uint4*)(pa + aK0);
    ua[1] = *(const uint4*)(pa + aK0 + 16);
#pragma unroll
    for (int j = 0; j < 2; j++) {
      v16h bfg; uint4* ub = (uint4*)&bfg;
      const _Float16* pb = &ks[nbase + j * 16 + rl][kt];
      ub[0] = *(const uint4*)(pb + bK0);
      ub[1] = *(const uint4*)(pb + bK0 + 16);
      acc[j] = WMMA_F16(af, bfg, acc[j]);
    }
  }
  float asum = 0.f;
  const float* rp = rpe + (size_t)(y * 8 + h) * 4096;
  float* Sp = S + (size_t)wh * 4096;
  const int mo = (lane >> 4) * 8;
#pragma unroll
  for (int j = 0; j < 2; j++) {
#pragma unroll
    for (int r = 0; r < 8; r++) {
      int m = mbase + mo + r, n = nbase + j * 16 + rl;
      float s = acc[j][r] + kBIAS * rp[m * 64 + n];
      Sp[m * 64 + n] = s;
      asum += fabsf(s);
    }
  }
  red[tid] = asum;
  __syncthreads();
  for (int s = 128; s > 0; s >>= 1) {
    if (tid < s) red[tid] += red[tid + s];
    __syncthreads();
  }
  if (tid == 0) partial[wh] = red[0];
}

// ---------------------------------------------------------------------------
// Pass B: M[w] = max( mean|attn_w| / max_w, 0.5 )   (256 windows, one block)
// ---------------------------------------------------------------------------
__global__ __launch_bounds__(256) void attn_m_kernel(const float* __restrict__ partial,
                                                     float* __restrict__ Mout) {
  __shared__ float red[256];
  int w = threadIdx.x;
  float s = 0.f;
  for (int h = 0; h < 8; h++) s += partial[w * 8 + h];
  float mw = s * (1.f / 32768.f);  // mean over HEADS*64*64
  red[w] = mw;
  __syncthreads();
  for (int st = 128; st > 0; st >>= 1) {
    if (w < st) red[w] = fmaxf(red[w], red[w + st]);
    __syncthreads();
  }
  Mout[w] = fmaxf(mw / red[0], 0.5f);
}

// ---------------------------------------------------------------------------
// Pass C: row-normalized cosine nonlinearity + O = P @ V, fused residual
//   P[q,k] = (1 - cos(S/||S_row|| * pi/2)) * M_w  (f16 in LDS)
//   att[b,l,c] = O + x * (1 - M_w)
// ---------------------------------------------------------------------------
__global__ __launch_bounds__(256) void attn_ov_kernel(const float* __restrict__ S,
                                                      const _Float16* __restrict__ vT,
                                                      const float* __restrict__ Mbuf,
                                                      const float* __restrict__ x,
                                                      _Float16* __restrict__ att) {
  __shared__ _Float16 Ps[64][72];
  __shared__ _Float16 Vs[64][72];
  const int wh = blockIdx.x, w = wh >> 3, h = wh & 7;
  const int tid = threadIdx.x, lane = tid & 31, wave = tid >> 5;
  const float Mw = Mbuf[w];
  const float* Sp = S + (size_t)wh * 4096;
  const _Float16* vp = vT + (size_t)wh * 4096;
#pragma unroll
  for (int i = 0; i < 2; i++) {
    int chunk = tid + i * 256;
    int row = chunk >> 3, cc = (chunk & 7) * 8;
    *(uint4*)&Vs[row][cc] = *(const uint4*)&vp[row * 64 + cc];
  }
  // row norms over k (64 elems/row): 4 threads per row, shuffle-combine
  const int r = tid >> 2, seg = (tid & 3) * 16;
  float sv[16];
  float ss = 0.f;
#pragma unroll
  for (int i = 0; i < 16; i++) {
    float t = Sp[r * 64 + seg + i];
    sv[i] = t;
    ss += t * t;
  }
  ss += __shfl_xor(ss, 1, 32);
  ss += __shfl_xor(ss, 2, 32);
  float nrm = fmaxf(sqrtf(ss), 1e-12f);
  float sc = kPI_2 / nrm;
#pragma unroll
  for (int i = 0; i < 16; i++)
    Ps[r][seg + i] = (_Float16)((1.f - cosf(sv[i] * sc)) * Mw);
  __syncthreads();

  const int mbase = (wave >> 1) * 16, nbase = (wave & 1) * 32;
  const int rl = lane & 15, aK0 = (lane >> 4) * 8, bK0 = (lane >> 4) * 8;
  v8f zero = {};
  v8f acc[2];
  acc[0] = zero; acc[1] = zero;
#pragma unroll
  for (int kt = 0; kt < 64; kt += 32) {
    v16h af; uint4* ua = (uint4*)&af;
    const _Float16* pa = &Ps[mbase + rl][kt];
    ua[0] = *(const uint4*)(pa + aK0);
    ua[1] = *(const uint4*)(pa + aK0 + 16);
#pragma unroll
    for (int j = 0; j < 2; j++) {
      v16h bfg; uint4* ub = (uint4*)&bfg;
      const _Float16* pb = &Vs[nbase + j * 16 + rl][kt];  // rows of v^T (n = d)
      ub[0] = *(const uint4*)(pb + bK0);
      ub[1] = *(const uint4*)(pb + bK0 + 16);
      acc[j] = WMMA_F16(af, bfg, acc[j]);
    }
  }
  const int bb = w >> 6, y = w & 63;
  const int mo = (lane >> 4) * 8;
  const float om = 1.f - Mw;
#pragma unroll
  for (int j = 0; j < 2; j++) {
#pragma unroll
    for (int rr = 0; rr < 8; rr++) {
      int m = mbase + mo + rr;        // xw
      int n = nbase + j * 16 + rl;    // d
      size_t idx = ((size_t)bb * 4096 + y * 64 + m) * 512 + h * 64 + n;
      att[idx] = (_Float16)(acc[j][rr] + x[idx] * om);
    }
  }
}

// ---------------------------------------------------------------------------
// Host-side orchestration
// ---------------------------------------------------------------------------
extern "C" void kernel_launch(void* const* d_in, const int* in_sizes, int n_in,
                              void* d_out, int out_size, void* d_ws, size_t ws_size,
                              hipStream_t stream) {
  const float* x       = (const float*)d_in[0];
  const float* rpe     = (const float*)d_in[1];
  const float* qkv_w   = (const float*)d_in[2];
  const float* proj_w  = (const float*)d_in[3];
  const float* proj_b  = (const float*)d_in[4];
  const float* norm1_w = (const float*)d_in[5];
  const float* norm1_b = (const float*)d_in[6];
  const float* norm2_w = (const float*)d_in[7];
  const float* norm2_b = (const float*)d_in[8];
  const float* fc1_w   = (const float*)d_in[9];
  const float* fc1_b   = (const float*)d_in[10];
  const float* fc2_w   = (const float*)d_in[11];
  const float* fc2_b   = (const float*)d_in[12];
  float* out = (float*)d_out;

  const size_t MB = 1u << 20;
  char* ws = (char*)d_ws;
  // Workspace layout (regions reused in dependency order; peak ~152 MB)
  _Float16* wqkv  = (_Float16*)(ws + 0);                 // 1536*512 f16
  _Float16* wproj = (_Float16*)(ws + 1572864);           // 512*512
  _Float16* wfc1  = (_Float16*)(ws + 2097152);           // 2048*512
  _Float16* wfc2  = (_Float16*)(ws + 4194304);           // 512*2048
  _Float16* img   = (_Float16*)(ws + 8 * MB);            // 16 MB, dead after QKV GEMM
  _Float16* qkvo  = (_Float16*)(ws + 24 * MB);           // 48 MB, dead after repack
  _Float16* qbuf  = (_Float16*)(ws + 72 * MB);           // 16 MB
  _Float16* kbuf  = (_Float16*)(ws + 88 * MB);           // 16 MB
  _Float16* vtbuf = (_Float16*)(ws + 104 * MB);          // 16 MB
  float*    x1    = (float*)(ws + 120 * MB);             // 32 MB fp32
  float*    part  = (float*)(ws + 152 * MB);             // 8 KB
  float*    Mbuf  = (float*)(ws + 152 * MB + 8192);      // 1 KB
  float*    Sbuf  = (float*)(ws + 24 * MB);              // 33.5 MB (reuse qkvo)
  _Float16* attb  = (_Float16*)(ws + 8 * MB);            // 16 MB (reuse img)
  _Float16* ln2b  = (_Float16*)(ws + 88 * MB);           // 16 MB (reuse kbuf)
  _Float16* gelu  = (_Float16*)(ws + 24 * MB);           // 64 MB (reuse qkvo/Sbuf/qbuf)

  // 1) weights -> f16
  cvt_kernel<<<(1536 * 512 + 255) / 256, 256, 0, stream>>>(qkv_w, wqkv, 1536 * 512);
  cvt_kernel<<<(512 * 512 + 255) / 256, 256, 0, stream>>>(proj_w, wproj, 512 * 512);
  cvt_kernel<<<(2048 * 512 + 255) / 256, 256, 0, stream>>>(fc1_w, wfc1, 2048 * 512);
  cvt_kernel<<<(512 * 2048 + 255) / 256, 256, 0, stream>>>(fc2_w, wfc2, 512 * 2048);
  // 2) LN1
  ln_kernel<<<kM, 256, 0, stream>>>(x, norm1_w, norm1_b, img);
  // 3) QKV GEMM (16384 x 1536 x 512)
  gemm_kernel<0><<<dim3(12, 128), 256, 0, stream>>>(img, wqkv, nullptr, nullptr,
                                                    qkvo, nullptr, kM, 1536, 512);
  // 4) repack into q/k/v^T window-head tiles
  repack_kernel<<<(kM * 512) / 256, 256, 0, stream>>>(qkvo, qbuf, kbuf, vtbuf);
  // 5) S = q k^T + bias ; partial |S| sums
  attn_qk_kernel<<<kWH, 256, 0, stream>>>(qbuf, kbuf, rpe, Sbuf, part);
  // 6) global M statistic
  attn_m_kernel<<<1, 256, 0, stream>>>(part, Mbuf);
  // 7) cosine nonlinearity + O = P V + residual x(1-M)
  attn_ov_kernel<<<kWH, 256, 0, stream>>>(Sbuf, vtbuf, Mbuf, x, attb);
  // 8) proj GEMM, x1 = x + attened @ Wp^T + bp
  gemm_kernel<1><<<dim3(4, 128), 256, 0, stream>>>(attb, wproj, proj_b, x,
                                                   nullptr, x1, kM, 512, 512);
  // 9) LN2
  ln_kernel<<<kM, 256, 0, stream>>>(x1, norm2_w, norm2_b, ln2b);
  // 10) FC1 + exact GELU (16384 x 2048 x 512)
  gemm_kernel<2><<<dim3(16, 128), 256, 0, stream>>>(ln2b, wfc1, fc1_b, nullptr,
                                                    gelu, nullptr, kM, 2048, 512);
  // 11) FC2 + residual x1 -> out (16384 x 512 x 2048)
  gemm_kernel<1><<<dim3(4, 128), 256, 0, stream>>>(gelu, wfc2, fc2_b, x1,
                                                   nullptr, out, kM, 512, 2048);
}
